// LatentEncoder_76192719831698
// MI455X (gfx1250) — compile-verified, matrix operand order checked by reading, not run on hardware
//
#include <hip/hip_runtime.h>
#include <hip/hip_bf16.h>

#define NNODE 16384
#define KNN   30
#define NEDGE (NNODE*KNN)

typedef __attribute__((ext_vector_type(16))) _Float16 v16h;
typedef __attribute__((ext_vector_type(4)))  _Float16 v4h;
typedef __attribute__((ext_vector_type(8)))  float    v8f;
typedef _Float16 h16;

__device__ __forceinline__ float sigm_(float x){ return 1.f/(1.f+__expf(-x)); }
__device__ __forceinline__ float silu_(float x){ return x*sigm_(x); }
__device__ __forceinline__ float fixnum_(float x){ return (x==x)?x:0.f; }

// ---- WMMA fragment helpers (CDNA5 16x16x32 f16 layout) -------------------
template<typename F>
__device__ __forceinline__ v16h make_frag(int lane, F f){
  const int r  = lane & 15;
  const int kb = (lane & 16) ? 8 : 0;
  v16h x;
#pragma unroll
  for (int q=0;q<8;++q){
    int k = (q<4) ? (2*q+kb) : (16+2*(q-4)+kb);
    x[2*q]   = f(r,k);
    x[2*q+1] = f(r,k+1);
  }
  return x;
}
template<typename F>
__device__ __forceinline__ void store_acc(int lane, v8f c, F f){
  const int col = lane & 15;
  const int mb  = (lane & 16) ? 8 : 0;
#pragma unroll
  for (int q=0;q<8;++q) f(mb+q, col, c[q]);
}
__device__ __forceinline__ v8f wmma16(v16h a, v16h b, v8f c){
  return __builtin_amdgcn_wmma_f32_16x16x32_f16(false,a,false,b,(short)0,c,false,false);
}
// packed B fragment: ((frag*32+lane)*16) halves, contiguous 32B per lane
__device__ __forceinline__ v16h load_pk(const h16* __restrict__ pk, int frag, int lane){
  return *(const v16h*)(pk + (((size_t)frag<<5) + lane)*16);
}

// ---------------------------------------------------------------------------
// K0: pack weight matrix into WMMA B-fragment layout (f16, zero-padded).
// virtual source: rows [0,split) from W0, [split,Krows) from W1, cols < Nsrc.
// ---------------------------------------------------------------------------
__global__ void k_pack_w(const float* __restrict__ W0, const float* __restrict__ W1,
                         int split, int Krows, int Nsrc, int KP, int NP,
                         h16* __restrict__ out)
{
  int idx = blockIdx.x*blockDim.x + threadIdx.x;
  if (idx >= KP*NP) return;
  int e    = idx & 15;
  int lane = (idx >> 4) & 31;
  int frag = idx >> 9;
  int NT = NP/16;
  int kc = frag / NT, nt = frag % NT;
  int col = lane & 15;
  int kb  = (lane & 16) ? 8 : 0;
  int q = e >> 1;
  int k = ((q<4) ? (2*q+kb) : (16+2*(q-4)+kb)) + (e & 1);
  int kk = kc*32 + k;
  int n  = nt*16 + col;
  float v = 0.f;
  if (n < Nsrc && kk < Krows){
    v = (kk < split) ? W0[kk*Nsrc + n] : W1[(kk-split)*Nsrc + n];
  }
  out[idx] = (h16)v;
}

// ---------------------------------------------------------------------------
// K1: backbone node features -> bbfeat [N][4][8]
// ---------------------------------------------------------------------------
__global__ void k_node_feats(const float* __restrict__ bb, float* __restrict__ bf)
{
  int i = blockIdx.x*blockDim.x + threadIdx.x;
  if (i>=NNODE) return;
  auto atom = [&](int a, float* p){ int nd=a/3, s=a%3; const float* q=bb+nd*12+s*3;
                                    p[0]=q[0]; p[1]=q[1]; p[2]=q[2]; };
  auto nrm3 = [&](float* v){ float l=sqrtf(v[0]*v[0]+v[1]*v[1]+v[2]*v[2])+1e-8f;
                             v[0]/=l; v[1]/=l; v[2]/=l; };
  float D[3];
#pragma unroll
  for (int t=0;t<3;++t){
    int j = 3*i + t - 1;
    if (j<0 || j>3*NNODE-4){ D[t]=0.f; continue; }
    float X0[3],X1[3],X2[3],X3[3];
    atom(j,X0); atom(j+1,X1); atom(j+2,X2); atom(j+3,X3);
    float u2[3],u1[3],u0[3];
    for(int a=0;a<3;a++){ u2[a]=X1[a]-X0[a]; u1[a]=X2[a]-X1[a]; u0[a]=X3[a]-X2[a]; }
    nrm3(u2); nrm3(u1); nrm3(u0);
    float n2[3]={u2[1]*u1[2]-u2[2]*u1[1], u2[2]*u1[0]-u2[0]*u1[2], u2[0]*u1[1]-u2[1]*u1[0]};
    float n1[3]={u1[1]*u0[2]-u1[2]*u0[1], u1[2]*u0[0]-u1[0]*u0[2], u1[0]*u0[1]-u1[1]*u0[0]};
    nrm3(n2); nrm3(n1);
    float cd = n2[0]*n1[0]+n2[1]*n1[1]+n2[2]*n1[2];
    cd = fminf(fmaxf(cd,-1.f+1e-7f),1.f-1e-7f);
    float sg = u2[0]*n1[0]+u2[1]*n1[1]+u2[2]*n1[2];
    float s  = (sg>0.f)?1.f:((sg<0.f)?-1.f:0.f);
    D[t] = s*acosf(cd);
  }
  float ca[3]={bb[i*12+3],bb[i*12+4],bb[i*12+5]};
  float fvec[3]={0,0,0}, bvec[3]={0,0,0};
  if (i < NNODE-1){
    float nx[3]={bb[(i+1)*12+3]-ca[0], bb[(i+1)*12+4]-ca[1], bb[(i+1)*12+5]-ca[2]};
    nrm3(nx); fvec[0]=nx[0]; fvec[1]=nx[1]; fvec[2]=nx[2];
  }
  if (i > 0){
    float px[3]={ca[0]-bb[(i-1)*12+3], ca[1]-bb[(i-1)*12+4], ca[2]-bb[(i-1)*12+5]};
    nrm3(px); bvec[0]=-px[0]; bvec[1]=-px[1]; bvec[2]=-px[2];
  }
  float Na[3]={bb[i*12+0],bb[i*12+1],bb[i*12+2]};
  float Cc[3]={bb[i*12+6],bb[i*12+7],bb[i*12+8]};
  float b_[3]={ca[0]-Na[0],ca[1]-Na[1],ca[2]-Na[2]};
  float c_[3]={Cc[0]-ca[0],Cc[1]-ca[1],Cc[2]-ca[2]};
  float a_[3]={b_[1]*c_[2]-b_[2]*c_[1], b_[2]*c_[0]-b_[0]*c_[2], b_[0]*c_[1]-b_[1]*c_[0]};
  float cb[3];
  for (int a=0;a<3;a++) cb[a] = -0.58273431f*a_[a] + 0.56802827f*b_[a] - 0.54067466f*c_[a] + ca[a];
  float* o = bf + i*32;
  o[0]=__cosf(D[0]); o[1]=__cosf(D[1]); o[2]=__cosf(D[2]);
  o[3]=__sinf(D[0]); o[4]=__sinf(D[1]); o[5]=__sinf(D[2]);
  o[6]=0.f; o[7]=0.f;
  for (int ch=0; ch<7; ++ch){
    float vx,vy,vz;
    if (ch<4){ vx=bb[i*12+ch*3+0]-ca[0]; vy=bb[i*12+ch*3+1]-ca[1]; vz=bb[i*12+ch*3+2]-ca[2]; }
    else if (ch==4){ vx=bvec[0]; vy=bvec[1]; vz=bvec[2]; }
    else if (ch==5){ vx=fvec[0]; vy=fvec[1]; vz=fvec[2]; }
    else { vx=cb[0]; vy=cb[1]; vz=cb[2]; }
    o[8+ch]=fixnum_(vx); o[16+ch]=fixnum_(vy); o[24+ch]=fixnum_(vz);
  }
  o[15]=0.f; o[23]=0.f; o[31]=0.f;
}

// ---------------------------------------------------------------------------
// K1b: atom node features -> atfeat [N][4][96]
// ---------------------------------------------------------------------------
__global__ void k_atom_feats(const float* __restrict__ atom91, const int* __restrict__ seq,
                             const unsigned char* __restrict__ noise, float* __restrict__ af)
{
  int idx = blockIdx.x*blockDim.x + threadIdx.x;
  if (idx >= NNODE*96) return;
  int n = idx/96, ch = idx%96;
  float v0=0,v1=0,v2=0,v3=0;
  if (ch < 20){
    float cond = (!noise[n] && seq[n]==ch) ? 1.f : 0.f;
    v0=v1=v2=v3=cond;
  } else if (ch < 91){
    const float* p = atom91 + n*273 + ch*3;
    v1=p[0]; v2=p[1]; v3=p[2];
  }
  float* o = af + n*384 + ch;
  o[0]=v0; o[96]=v1; o[192]=v2; o[288]=v3;
}

// ---------------------------------------------------------------------------
// K2: KNN — 30 rounds of block argmin with LDS bitmask exclusion.
// ---------------------------------------------------------------------------
__global__ void __launch_bounds__(256) k_knn(const float* __restrict__ bb,
                                             const unsigned char* __restrict__ xmask,
                                             const int* __restrict__ batch,
                                             int* __restrict__ srcidx)
{
  __shared__ float    s_d[256];
  __shared__ int      s_i[256];
  __shared__ unsigned s_mask[NNODE/32];
  int n = blockIdx.x, tid = threadIdx.x;
  for (int idx=tid; idx<NNODE/32; idx+=256) s_mask[idx]=0u;
  int bn = batch[n];
  bool mn = xmask[n]!=0;
  float cx = mn?1e9f:bb[n*12+3], cy = mn?1e9f:bb[n*12+4], cz = mn?1e9f:bb[n*12+5];
  __syncthreads();
  for (int r=0;r<KNN;++r){
    float best=__builtin_inff(); int bj=-1;
    for (int j=tid;j<NNODE;j+=256){
      if (j==n) continue;
      if (batch[j]!=bn) continue;
      if ((s_mask[j>>5]>>(j&31))&1u) continue;
      bool mj = xmask[j]!=0;
      float dx=cx-(mj?1e9f:bb[j*12+3]);
      float dy=cy-(mj?1e9f:bb[j*12+4]);
      float dz=cz-(mj?1e9f:bb[j*12+5]);
      float d2=dx*dx+dy*dy+dz*dz;
      if (d2<best || (d2==best && j<bj)){ best=d2; bj=j; }
    }
    s_d[tid]=best; s_i[tid]=bj;
    __syncthreads();
    for (int s=128;s>0;s>>=1){
      if (tid<s){
        if (s_i[tid+s]>=0 && (s_d[tid+s]<s_d[tid] ||
            (s_d[tid+s]==s_d[tid] && (s_i[tid]<0 || s_i[tid+s]<s_i[tid])))){
          s_d[tid]=s_d[tid+s]; s_i[tid]=s_i[tid+s];
        }
      }
      __syncthreads();
    }
    if (tid==0){
      int w=s_i[0]<0?0:s_i[0];
      s_mask[w>>5] |= 1u<<(w&31);
      srcidx[n*KNN+r]=w;
    }
    __syncthreads();
  }
}

// ---------------------------------------------------------------------------
// K3: edge rotation matrices + edge features
// ---------------------------------------------------------------------------
__global__ void k_edge_feats(const float* __restrict__ bb, const int* __restrict__ srcidx,
                             float* __restrict__ Rm, float* __restrict__ ef)
{
  int e = blockIdx.x*blockDim.x + threadIdx.x;
  if (e>=NEDGE) return;
  int dst = e/KNN, src = srcidx[e];
  float ex = bb[dst*12+3]-bb[src*12+3];
  float ey = bb[dst*12+4]-bb[src*12+4];
  float ez = bb[dst*12+5]-bb[src*12+5];
  float d  = sqrtf(ex*ex+ey*ey+ez*ez);
  float inv = 1.f/(d+1e-8f);
  float yx=ex*inv, yy=ey*inv, yz=ez*inv;
  float hx,hy,hz;
  if (fabsf(yy)<0.99f){hx=0;hy=1;hz=0;} else {hx=1;hy=0;hz=0;}
  float xx=hy*yz-hz*yy, xy=hz*yx-hx*yz, xz=hx*yy-hy*yx;
  float xl=sqrtf(xx*xx+xy*xy+xz*xz)+1e-8f; xx/=xl; xy/=xl; xz/=xl;
  float zx=xy*yz-xz*yy, zy=xz*yx-xx*yz, zz=xx*yy-xy*yx;
  float* R = Rm + e*9;
  R[0]=xx; R[1]=xy; R[2]=xz;
  R[3]=yx; R[4]=yy; R[5]=yz;
  R[6]=zx; R[7]=zy; R[8]=zz;
  float* E = ef + e*32;
#pragma unroll
  for (int i=0;i<16;++i){
    float mu = (20.f/15.f)*(float)i;
    float t  = (d-mu)*(1.f/1.25f);
    E[i] = __expf(-t*t);
  }
  float dd = (float)(src-dst);
#pragma unroll
  for (int i=0;i<8;++i){
    float fr = __expf(-(float)(2*i)*0.57564627325f);
    float an = dd*fr;
    E[16+i]=__cosf(an); E[24+i]=__sinf(an);
  }
}

// ---------------------------------------------------------------------------
// K4: so3_conv embed. Packed-f16 weights, float4 gathers, WMMA GEMMs.
// ---------------------------------------------------------------------------
template<int CIN, int CINP, int K2P>
__global__ void __launch_bounds__(128) k_so3conv(
  const float* __restrict__ feat, const float* __restrict__ ef,
  const float* __restrict__ Rm, const int* __restrict__ srcidx,
  const h16* __restrict__ pk_wrad1, const float* __restrict__ b_rad1,
  const h16* __restrict__ pk_wrad2, const float* __restrict__ b_rad2,
  const h16* __restrict__ pk_W2,
  float* __restrict__ xout, int out_off)
{
  constexpr int NT2 = K2P/16;
  constexpr int KC  = K2P/32;
  constexpr int C4  = CINP/4;
  __shared__ __align__(16) h16   s_xs[32][4][CINP];
  __shared__ __align__(16) h16   s_ef[32][32];
  __shared__ __align__(16) h16   s_rad[32][K2P];
  __shared__ __align__(16) h16   s_h[32][64];
  __shared__ float s_R[32][9];
  __shared__ float s_msg[4][16][32];
  __shared__ float s_acc[4][4][32];

  int n = blockIdx.x, tid = threadIdx.x;
  int wave = tid>>5, lane = tid&31;

  for (int idx=tid; idx<32*9; idx+=128){
    int j=idx/9, q=idx%9;
    s_R[j][q] = (j<KNN) ? Rm[(size_t)(n*KNN+j)*9+q] : 0.f;
  }
  for (int idx=tid; idx<4*4*32; idx+=128) ((float*)s_acc)[idx]=0.f;
  // ef tile (f16)
  for (int idx=tid; idx<32*8; idx+=128){
    int j=idx>>3, c4=idx&7;
    v4h o={0,0,0,0};
    if (j<KNN){
      float4 a = *((const float4*)(ef + (size_t)(n*KNN+j)*32) + c4);
      o[0]=(h16)a.x; o[1]=(h16)a.y; o[2]=(h16)a.z; o[3]=(h16)a.w;
    }
    *(v4h*)&s_ef[j][c4*4]=o;
  }
  __syncthreads();

  // gather + rot_fwd (float4 loads; feat buffers are zero padded to CINP)
  for (int idx=tid; idx<32*C4; idx+=128){
    int j=idx/C4, c4=idx%C4;
    v4h o0={0,0,0,0}, o1=o0, o2=o0, o3=o0;
    if (j<KNN){
      int s = srcidx[n*KNN+j];
      const float4* fp = (const float4*)(feat + (size_t)s*4*CINP) + c4;
      float4 a0=fp[0], a1=fp[C4], a2=fp[2*C4], a3=fp[3*C4];
      float r0=s_R[j][0],r1=s_R[j][1],r2=s_R[j][2],r3=s_R[j][3],r4=s_R[j][4],
            r5=s_R[j][5],r6=s_R[j][6],r7=s_R[j][7],r8=s_R[j][8];
      float b0[4]={a0.x,a0.y,a0.z,a0.w};
      float b1[4]={a1.x,a1.y,a1.z,a1.w};
      float b2[4]={a2.x,a2.y,a2.z,a2.w};
      float b3[4]={a3.x,a3.y,a3.z,a3.w};
#pragma unroll
      for (int i=0;i<4;++i){
        o0[i]=(h16)b0[i];
        o1[i]=(h16)(r0*b1[i]+r1*b2[i]+r2*b3[i]);
        o2[i]=(h16)(r3*b1[i]+r4*b2[i]+r5*b3[i]);
        o3[i]=(h16)(r6*b1[i]+r7*b2[i]+r8*b3[i]);
      }
    }
    *(v4h*)&s_xs[j][0][c4*4]=o0;
    *(v4h*)&s_xs[j][1][c4*4]=o1;
    *(v4h*)&s_xs[j][2][c4*4]=o2;
    *(v4h*)&s_xs[j][3][c4*4]=o3;
  }
  __syncthreads();

  // rad stage 1: h = silu(ef @ w_rad1 + b_rad1)  rows=32, K=32, N=64
  for (int jid=wave; jid<8; jid+=4){
    int mt = jid & 1, nt = jid >> 1;
    v8f acc = {};
    v16h A = make_frag(lane,[&](int r,int k)->h16{ return s_ef[mt*16+r][k]; });
    v16h B = load_pk(pk_wrad1, nt, lane);
    acc = wmma16(A,B,acc);
    store_acc(lane,acc,[&](int m,int col_,float v){
      int col = nt*16+col_;
      s_h[mt*16+m][col] = (h16)silu_(v + b_rad1[col]);
    });
  }
  __syncthreads();

  // rad stage 2: rad = h @ w_rad2 + b_rad2  rows=32, K=64, N=K2P
  for (int jid=wave; jid<2*NT2; jid+=4){
    int mt = jid % 2, nt = jid / 2;
    v8f acc = {};
#pragma unroll
    for (int kc=0;kc<2;++kc){
      v16h A = make_frag(lane,[&](int r,int k)->h16{ return s_h[mt*16+r][kc*32+k]; });
      v16h B = load_pk(pk_wrad2, kc*NT2+nt, lane);
      acc = wmma16(A,B,acc);
    }
    store_acc(lane,acc,[&](int m,int col_,float v){
      int c = nt*16+col_;
      float bv = (c<2*CIN)? b_rad2[c] : 0.f;
      s_rad[mt*16+m][c] = (h16)(v+bv);
    });
  }
  __syncthreads();

  // main GEMM: rows=(edge,comp), A = xs*rad, W=[w0;w1], N=32
  for (int g=wave; g<8; g+=4){
#pragma unroll
    for (int nt=0; nt<2; ++nt){
      v8f acc = {};
      for (int kc=0;kc<KC;++kc){
        v16h A = make_frag(lane,[&](int r,int k)->h16{
          int kk = kc*32+k;
          int j = g*4 + (r>>2), c = r&3;
          if (kk < CIN)        return (c==0) ? (h16)(s_xs[j][0][kk]     * s_rad[j][kk]) : (h16)0.f;
          else if (kk < 2*CIN) return (c!=0) ? (h16)(s_xs[j][c][kk-CIN] * s_rad[j][kk]) : (h16)0.f;
          return (h16)0.f;
        });
        v16h B = load_pk(pk_W2, kc*2+nt, lane);
        acc = wmma16(A,B,acc);
      }
      store_acc(lane,acc,[&](int m,int col_,float v){ s_msg[wave][m][nt*16+col_] = v; });
    }
#pragma unroll
    for (int le=0; le<4; ++le){
      int j = g*4+le;
      if (j>=KNN) continue;
      float m0 = s_msg[wave][le*4+0][lane];
      float m1 = s_msg[wave][le*4+1][lane];
      float m2 = s_msg[wave][le*4+2][lane];
      float m3 = s_msg[wave][le*4+3][lane];
      s_acc[wave][0][lane] += m0;
      s_acc[wave][1][lane] += s_R[j][0]*m1 + s_R[j][3]*m2 + s_R[j][6]*m3;
      s_acc[wave][2][lane] += s_R[j][1]*m1 + s_R[j][4]*m2 + s_R[j][7]*m3;
      s_acc[wave][3][lane] += s_R[j][2]*m1 + s_R[j][5]*m2 + s_R[j][8]*m3;
    }
  }
  __syncthreads();
  {
    int c = tid>>5, col = tid&31;
    float s = s_acc[0][c][col]+s_acc[1][c][col]+s_acc[2][c][col]+s_acc[3][c][col];
    xout[(size_t)n*256 + c*64 + out_off + col] = s * (1.f/(float)KNN);
  }
}

// ---------------------------------------------------------------------------
// K5: attention block.
// ---------------------------------------------------------------------------
template<int COUT>
__global__ void __launch_bounds__(128) k_attn(
  const float* __restrict__ xin, const float* __restrict__ ef,
  const float* __restrict__ Rm, const int* __restrict__ srcidx,
  const h16* __restrict__ pk_wa1, const float* __restrict__ ba1,
  const h16* __restrict__ pk_wa2, const h16* __restrict__ pk_wv,
  const float* __restrict__ wo, const float* __restrict__ wf1,
  const float* __restrict__ wf2, float* __restrict__ xout)
{
  __shared__ __align__(16) h16   s_xs[32][4][64];
  __shared__ __align__(16) h16   s_ef[32][32];
  __shared__ h16   s_xd[64];
  __shared__ float s_R[32][9];
  __shared__ __align__(16) h16   s_a1[32][32];
  __shared__ float s_ar[32][8];
  __shared__ float s_alpha[32][8];
  __shared__ float s_v[32][4][64];
  __shared__ float s_out[4][64];
  __shared__ float s_y[4][COUT];
  __shared__ float s_g[4][32];

  int n = blockIdx.x, tid=threadIdx.x, wave=tid>>5, lane=tid&31;

  for (int idx=tid; idx<32*9; idx+=128){
    int j=idx/9, q=idx%9;
    s_R[j][q] = (j<KNN) ? Rm[(size_t)(n*KNN+j)*9+q] : 0.f;
  }
  if (tid<64) s_xd[tid] = (h16)xin[(size_t)n*256 + tid];
  for (int idx=tid; idx<32*8; idx+=128){
    int j=idx>>3, c4=idx&7;
    v4h o={0,0,0,0};
    if (j<KNN){
      float4 a = *((const float4*)(ef + (size_t)(n*KNN+j)*32) + c4);
      o[0]=(h16)a.x; o[1]=(h16)a.y; o[2]=(h16)a.z; o[3]=(h16)a.w;
    }
    *(v4h*)&s_ef[j][c4*4]=o;
  }
  __syncthreads();

  for (int idx=tid; idx<32*16; idx+=128){
    int j=idx>>4, c4=idx&15;
    v4h o0={0,0,0,0}, o1=o0, o2=o0, o3=o0;
    if (j<KNN){
      int s = srcidx[n*KNN+j];
      const float4* fp = (const float4*)(xin + (size_t)s*256) + c4;
      float4 a0=fp[0], a1=fp[16], a2=fp[32], a3=fp[48];
      float r0=s_R[j][0],r1=s_R[j][1],r2=s_R[j][2],r3=s_R[j][3],r4=s_R[j][4],
            r5=s_R[j][5],r6=s_R[j][6],r7=s_R[j][7],r8=s_R[j][8];
      float b0[4]={a0.x,a0.y,a0.z,a0.w};
      float b1[4]={a1.x,a1.y,a1.z,a1.w};
      float b2[4]={a2.x,a2.y,a2.z,a2.w};
      float b3[4]={a3.x,a3.y,a3.z,a3.w};
#pragma unroll
      for (int i=0;i<4;++i){
        o0[i]=(h16)b0[i];
        o1[i]=(h16)(r0*b1[i]+r1*b2[i]+r2*b3[i]);
        o2[i]=(h16)(r3*b1[i]+r4*b2[i]+r5*b3[i]);
        o3[i]=(h16)(r6*b1[i]+r7*b2[i]+r8*b3[i]);
      }
    }
    *(v4h*)&s_xs[j][0][c4*4]=o0;
    *(v4h*)&s_xs[j][1][c4*4]=o1;
    *(v4h*)&s_xs[j][2][c4*4]=o2;
    *(v4h*)&s_xs[j][3][c4*4]=o3;
  }
  __syncthreads();

  // a1 = silu(inv @ wa1 + ba1): rows=32, K=160, N=32
  {
    int mt = wave&1, nt = wave>>1;
    v8f acc = {};
#pragma unroll
    for (int kc=0;kc<5;++kc){
      v16h A = make_frag(lane,[&](int r,int k)->h16{
        int j = mt*16+r, kk = kc*32+k;
        if (j>=KNN) return (h16)0.f;
        if (kk<64)  return s_xs[j][0][kk];
        if (kk<128) return s_xd[kk-64];
        return s_ef[j][kk-128];
      });
      v16h B = load_pk(pk_wa1, kc*2+nt, lane);
      acc = wmma16(A,B,acc);
    }
    store_acc(lane,acc,[&](int m,int col_,float v){
      int col = nt*16+col_;
      s_a1[mt*16+m][col] = (h16)silu_(v + ba1[col]);
    });
  }
  __syncthreads();

  // a = a1 @ wa2: rows=32, K=32, N=8(pad16)
  if (wave<2){
    v8f acc = {};
    v16h A = make_frag(lane,[&](int r,int k)->h16{ return s_a1[wave*16+r][k]; });
    v16h B = load_pk(pk_wa2, 0, lane);
    acc = wmma16(A,B,acc);
    store_acc(lane,acc,[&](int m,int col_,float v){
      if (col_<8) s_ar[wave*16+m][col_]=v;
    });
  }

  // v = rot_bwd(R, xs_rot @ wv): rows=128, K=64, N=64
  for (int g=wave; g<8; g+=4){
#pragma unroll
    for (int nt=0;nt<4;++nt){
      v8f acc = {};
#pragma unroll
      for (int kc=0;kc<2;++kc){
        v16h A = make_frag(lane,[&](int r,int k)->h16{
          int j = g*4+(r>>2), c = r&3;
          return s_xs[j][c][kc*32+k];
        });
        v16h B = load_pk(pk_wv, kc*4+nt, lane);
        acc = wmma16(A,B,acc);
      }
      store_acc(lane,acc,[&](int m,int col_,float v){
        int j = g*4+(m>>2), c = m&3;
        s_v[j][c][nt*16+col_] = v;
      });
    }
#pragma unroll
    for (int le=0;le<4;++le){
      int j = g*4+le;
      for (int cc=lane; cc<64; cc+=32){
        float m1=s_v[j][1][cc], m2=s_v[j][2][cc], m3=s_v[j][3][cc];
        s_v[j][1][cc]=s_R[j][0]*m1+s_R[j][3]*m2+s_R[j][6]*m3;
        s_v[j][2][cc]=s_R[j][1]*m1+s_R[j][4]*m2+s_R[j][7]*m3;
        s_v[j][3][cc]=s_R[j][2]*m1+s_R[j][5]*m2+s_R[j][8]*m3;
      }
    }
  }
  __syncthreads();

  if (tid<8){
    float mx=-__builtin_inff();
    for (int j=0;j<KNN;++j) mx=fmaxf(mx,s_ar[j][tid]);
    float sum=0.f;
    for (int j=0;j<KNN;++j){ float e=__expf(s_ar[j][tid]-mx); s_alpha[j][tid]=e; sum+=e; }
    float iv=1.f/sum;
    for (int j=0;j<KNN;++j) s_alpha[j][tid]*=iv;
  }
  __syncthreads();

  for (int idx=tid; idx<256; idx+=128){
    int c=idx>>6, col=idx&63, h=col>>3;
    float s=0.f;
    for (int j=0;j<KNN;++j) s += s_alpha[j][h]*s_v[j][c][col];
    s_out[c][col]=s;
  }
  __syncthreads();

  for (int idx=tid; idx<4*COUT; idx+=128){
    int c=idx/COUT, oc=idx%COUT;
    float s=0.f;
    for (int k=0;k<64;++k) s += s_out[c][k]*wo[k*COUT+oc];
    if (COUT==64) s += xin[(size_t)n*256 + c*64 + oc];
    s_y[c][oc]=s;
  }
  __syncthreads();

  {
    int c=tid>>5, hh=tid&31;
    float s=0.f;
    for (int k=0;k<COUT;++k) s += s_y[c][k]*wf1[k*32+hh];
    s_g[c][hh]=s;
  }
  __syncthreads();
  float gv;
  {
    int c=tid>>5, hh=tid&31;
    float h0=s_g[0][hh], hc=s_g[c][hh];
    gv = (c==0) ? silu_(hc) : hc*sigm_(h0);
  }
  __syncthreads();
  { int c=tid>>5, hh=tid&31; s_g[c][hh]=gv; }
  __syncthreads();
  for (int idx=tid; idx<4*COUT; idx+=128){
    int c=idx/COUT, oc=idx%COUT;
    float s=s_y[c][oc];
    for (int k=0;k<32;++k) s += s_g[c][k]*wf2[k*COUT+oc];
    xout[(size_t)n*4*COUT + c*COUT + oc] = s;
  }
}

// ---------------------------------------------------------------------------
// K6: edge update. ef tile staged with CDNA5 async-to-LDS; x gathers float4.
// ---------------------------------------------------------------------------
__global__ void __launch_bounds__(128) k_edge_update(
  const float* __restrict__ xin, float* __restrict__ ef,
  const int* __restrict__ srcidx,
  const h16* __restrict__ pk_we1, const float* __restrict__ be1,
  const h16* __restrict__ pk_we2)
{
  __shared__ __align__(16) h16   s_h[4][16][64];
  __shared__ __align__(16) float s_ef32[64][32];
  __shared__ __align__(16) h16   s_xg[64][128];
  int tid=threadIdx.x, wave=tid>>5, lane=tid&31;
  int e0 = blockIdx.x*64 + wave*16;

  // async copy of contiguous 8KB ef tile into LDS (CDNA5 ASYNC path)
  {
    unsigned lbase = (unsigned)(size_t)&s_ef32[0][0];
    unsigned long long g = (unsigned long long)(size_t)(ef + (size_t)blockIdx.x*64*32);
#pragma unroll
    for (int it=0; it<4; ++it){
      unsigned loff = lbase + (unsigned)((it*128 + tid)*16);
      unsigned long long ga = g + (unsigned long long)((it*128 + tid)*16);
      asm volatile("global_load_async_to_lds_b128 %0, %1, off"
                   :: "v"(loff), "v"(ga) : "memory");
    }
  }
  // gather x[src][0] / x[dst][0] rows while async copy is in flight
  for (int idx=tid; idx<64*32; idx+=128){
    int el=idx>>5, seg=idx&31;
    int e = blockIdx.x*64 + el;
    v4h o;
    if (seg<16){
      int s = srcidx[e];
      float4 a = *((const float4*)(xin + (size_t)s*256) + seg);
      o[0]=(h16)a.x; o[1]=(h16)a.y; o[2]=(h16)a.z; o[3]=(h16)a.w;
      *(v4h*)&s_xg[el][seg*4]=o;
    } else {
      int d = e/KNN;
      float4 a = *((const float4*)(xin + (size_t)d*256) + (seg-16));
      o[0]=(h16)a.x; o[1]=(h16)a.y; o[2]=(h16)a.z; o[3]=(h16)a.w;
      *(v4h*)&s_xg[el][64+(seg-16)*4]=o;
    }
  }
  asm volatile("s_wait_asynccnt 0" ::: "memory");
  __syncthreads();

#pragma unroll
  for (int nt=0; nt<4; ++nt){
    v8f acc = {};
#pragma unroll
    for (int kc=0;kc<5;++kc){
      v16h A = make_frag(lane,[&](int r,int k)->h16{
        int el = wave*16+r, kk = kc*32+k;
        if (kk<32) return (h16)s_ef32[el][kk];
        return s_xg[el][kk-32];
      });
      v16h B = load_pk(pk_we1, kc*4+nt, lane);
      acc = wmma16(A,B,acc);
    }
    store_acc(lane,acc,[&](int m,int col_,float v){
      int col = nt*16+col_;
      s_h[wave][m][col] = (h16)silu_(v + be1[col]);
    });
  }
#pragma unroll
  for (int nt=0;nt<2;++nt){
    v8f acc = {};
#pragma unroll
    for (int kc=0;kc<2;++kc){
      v16h A = make_frag(lane,[&](int r,int k)->h16{ return s_h[wave][r][kc*32+k]; });
      v16h B = load_pk(pk_we2, kc*2+nt, lane);
      acc = wmma16(A,B,acc);
    }
    store_acc(lane,acc,[&](int m,int col_,float v){
      int col = nt*16+col_, e = e0+m;
      ef[(size_t)e*32+col] += v;
    });
  }
}

// ---------------------------------------------------------------------------
extern "C" void kernel_launch(void* const* d_in, const int* in_sizes, int n_in,
                              void* d_out, int out_size, void* d_ws, size_t ws_size,
                              hipStream_t stream)
{
  (void)in_sizes; (void)n_in; (void)out_size; (void)ws_size;
  const float* bb      = (const float*)d_in[0];
  const float* atom91  = (const float*)d_in[1];
  const int*   seq     = (const int*)d_in[2];
  const unsigned char* xmask = (const unsigned char*)d_in[3];
  const int*   batch   = (const int*)d_in[4];
  const unsigned char* noise = (const unsigned char*)d_in[5];
  auto F = [&](int i){ return (const float*)d_in[i]; };
  const float *ea_brad1=F(6), *ea_brad2=F(7), *ea_w0=F(8), *ea_w1=F(9), *ea_wrad1=F(10), *ea_wrad2=F(11);
  const float *eb_brad1=F(12),*eb_brad2=F(13),*eb_w0=F(14),*eb_w1=F(15),*eb_wrad1=F(16),*eb_wrad2=F(17);

  char* w = (char*)d_ws;
  int*   srcidx = (int*)w;    w += (size_t)NEDGE*4;
  float* Rm     = (float*)w;  w += (size_t)NEDGE*9*4;
  float* efb    = (float*)w;  w += (size_t)NEDGE*32*4;
  float* bbfeat = (float*)w;  w += (size_t)NNODE*32*4;
  float* atfeat = (float*)w;  w += (size_t)NNODE*384*4;
  float* xa     = (float*)w;  w += (size_t)NNODE*256*4;
  float* xb     = (float*)w;  w += (size_t)NNODE*256*4;
  h16* pkcur    = (h16*)w;

  auto PACK = [&](const float* W0, const float* W1, int split, int Krows, int Nsrc,
                  int KP, int NP)->const h16*{
    h16* dst = pkcur;
    int total = KP*NP;
    k_pack_w<<<(total+255)/256, 256, 0, stream>>>(W0, W1, split, Krows, Nsrc, KP, NP, dst);
    pkcur += (size_t)total;
    return dst;
  };

  // embed weights
  const h16* pk_eb_wr1 = PACK(eb_wrad1, eb_wrad1, 32, 32, 64, 32, 64);
  const h16* pk_eb_wr2 = PACK(eb_wrad2, eb_wrad2, 64, 64, 14, 64, 32);
  const h16* pk_eb_W2  = PACK(eb_w0,    eb_w1,     7, 14, 32, 32, 32);
  const h16* pk_ea_wr1 = PACK(ea_wrad1, ea_wrad1, 32, 32, 64, 32, 64);
  const h16* pk_ea_wr2 = PACK(ea_wrad2, ea_wrad2, 64, 64, 182, 64, 192);
  const h16* pk_ea_W2  = PACK(ea_w0,    ea_w1,    91, 182, 32, 192, 32);
  // layer + head weights
  const h16 *pk_wa1[6], *pk_wa2[6], *pk_wv[6], *pk_we1[4], *pk_we2[4];
  for (int l=0; l<4; ++l){
    int b = 18 + 10*l;
    pk_wa1[l] = PACK(F(b+1), F(b+1), 160, 160, 32, 160, 32);
    pk_wa2[l] = PACK(F(b+2), F(b+2),  32,  32,  8,  32, 16);
    pk_wv[l]  = PACK(F(b+6), F(b+6),  64,  64, 64,  64, 64);
    pk_we1[l] = PACK(F(b+8), F(b+8), 160, 160, 64, 160, 64);
    pk_we2[l] = PACK(F(b+9), F(b+9),  64,  64, 32,  64, 32);
  }
  // out heads: logvar leaves at 58..64, mu at 65..71
  pk_wa1[4] = PACK(F(66), F(66), 160, 160, 32, 160, 32); // mu
  pk_wa2[4] = PACK(F(67), F(67),  32,  32,  8,  32, 16);
  pk_wv[4]  = PACK(F(69), F(69),  64,  64, 64,  64, 64);
  pk_wa1[5] = PACK(F(59), F(59), 160, 160, 32, 160, 32); // logvar
  pk_wa2[5] = PACK(F(60), F(60),  32,  32,  8,  32, 16);
  pk_wv[5]  = PACK(F(62), F(62),  64,  64, 64,  64, 64);

  k_node_feats<<<NNODE/256, 256, 0, stream>>>(bb, bbfeat);
  k_atom_feats<<<(NNODE*96)/256, 256, 0, stream>>>(atom91, seq, noise, atfeat);
  k_knn<<<NNODE, 256, 0, stream>>>(bb, xmask, batch, srcidx);
  k_edge_feats<<<NEDGE/256, 256, 0, stream>>>(bb, srcidx, Rm, efb);

  k_so3conv<7,8,32><<<NNODE, 128, 0, stream>>>(bbfeat, efb, Rm, srcidx,
      pk_eb_wr1, eb_brad1, pk_eb_wr2, eb_brad2, pk_eb_W2, xa, 0);
  k_so3conv<91,96,192><<<NNODE, 128, 0, stream>>>(atfeat, efb, Rm, srcidx,
      pk_ea_wr1, ea_brad1, pk_ea_wr2, ea_brad2, pk_ea_W2, xa, 32);

  float* xcur = xa; float* xnxt = xb;
  for (int l=0; l<4; ++l){
    int b = 18 + 10*l;
    k_attn<64><<<NNODE, 128, 0, stream>>>(xcur, efb, Rm, srcidx,
        pk_wa1[l], F(b+0), pk_wa2[l], pk_wv[l], F(b+5), F(b+3), F(b+4), xnxt);
    k_edge_update<<<NEDGE/64, 128, 0, stream>>>(xnxt, efb, srcidx,
        pk_we1[l], F(b+7), pk_we2[l]);
    float* t=xcur; xcur=xnxt; xnxt=t;
  }

  float* out = (float*)d_out;
  k_attn<32><<<NNODE, 128, 0, stream>>>(xcur, efb, Rm, srcidx,
      pk_wa1[4], F(65), pk_wa2[4], pk_wv[4], F(70), F(68), F(69), out);
  k_attn<32><<<NNODE, 128, 0, stream>>>(xcur, efb, Rm, srcidx,
      pk_wa1[5], F(58), pk_wa2[5], pk_wv[5], F(63), F(61), F(62),
      out + (size_t)NNODE*128);
}
